// Attention_45689862095051
// MI455X (gfx1250) — compile-verified, matrix operand order checked by reading, not run on hardware
//
#include <hip/hip_runtime.h>

// ---------------------------------------------------------------------------
// Problem constants
// ---------------------------------------------------------------------------
#define BATCH 4
#define SEQ   1024
#define DIM   4096
#define NH    32
#define HD    128
#define RANK  16
#define BS    (BATCH * SEQ)          // 4096 rows
#define LORA_SCALE 2.0f

// ---------------------------------------------------------------------------
// Types
// ---------------------------------------------------------------------------
typedef __bf16 bf16;
typedef __attribute__((ext_vector_type(16))) __bf16 bf16x16;
typedef __attribute__((ext_vector_type(8)))  __bf16 bf16x8;
typedef __attribute__((ext_vector_type(8)))  float  f32x8;
typedef __attribute__((ext_vector_type(4)))  float  f32x4;

__device__ __forceinline__ bf16 f2bf(float f) {
    unsigned u = __builtin_bit_cast(unsigned, f);
    unsigned r = u + 0x7FFFu + ((u >> 16) & 1u);   // round-to-nearest-even
    unsigned short h = (unsigned short)(r >> 16);
    return __builtin_bit_cast(__bf16, h);
}
__device__ __forceinline__ float bf2f(bf16 b) {
    unsigned short h = __builtin_bit_cast(unsigned short, b);
    unsigned u = ((unsigned)h) << 16;
    return __builtin_bit_cast(float, u);
}

// ---------------------------------------------------------------------------
// Async global->LDS copy (CDNA5 GLOBAL_LOAD_ASYNC_TO_LDS_B128, ASYNCcnt).
// Generic pointers to __shared__ carry the wave-relative LDS offset in their
// low 32 bits (ISA 10.2 aperture rules), which is exactly the VDST operand:
//   the LDS byte address is the LDS base plus that VGPR plus the instruction offset
// ---------------------------------------------------------------------------
__device__ __forceinline__ void async_b128(const void* gsrc, void* ldst) {
    unsigned l = (unsigned)(size_t)ldst;
    asm volatile("global_load_async_to_lds_b128 %0, %1, off"
                 :: "v"(l), "v"(gsrc) : "memory");
}

// ---------------------------------------------------------------------------
// WMMA fragment loader for row-major [row][k] tiles (LDS or global).
// CDNA5 16-bit A/B layout (CDNA5 ISA §7.12.2):
//   lane 0-15 : row = lane,      K = {0..7} then {16..23}
//   lane16-31 : row = lane-16,   K = {8..15} then {24..31}
// Two 16-byte loads per fragment.
// ---------------------------------------------------------------------------
__device__ __forceinline__ bf16x16 load_frag(const bf16* p, int ld, int r0, int k0) {
    int lane = threadIdx.x & 31;
    const bf16* row = p + (size_t)(r0 + (lane & 15)) * ld + k0 + ((lane >> 4) << 3);
    bf16x16 f;
    *(bf16x8*)&f       = *(const bf16x8*)(row);
    *((bf16x8*)&f + 1) = *(const bf16x8*)(row + 16);
    return f;
}

// ---------------------------------------------------------------------------
// fp32 -> bf16 bulk conversion (8 elems / thread, vectorized)
// ---------------------------------------------------------------------------
__global__ __launch_bounds__(256) void cvt_f32_bf16_kernel(
        const float* __restrict__ in, bf16* __restrict__ out, int n8) {
    int i = blockIdx.x * blockDim.x + threadIdx.x;
    if (i >= n8) return;
    const float* s = in + (size_t)i * 8;
    f32x4 a = *(const f32x4*)(s);
    f32x4 b = *(const f32x4*)(s + 4);
    bf16x8 o;
#pragma unroll
    for (int j = 0; j < 4; ++j) { o[j] = f2bf(a[j]); o[4 + j] = f2bf(b[j]); }
    *(bf16x8*)(out + (size_t)i * 8) = o;
}

// ---------------------------------------------------------------------------
// LoRA first stage: t[m][r] = sum_k x[m][k] * wa[r][k]   (fp32, rank 16, tiny)
// block = 256 threads = 16 rows x 16 ranks
// ---------------------------------------------------------------------------
__global__ __launch_bounds__(256) void lora_t_kernel(
        const float* __restrict__ x, const float* __restrict__ wa,
        float* __restrict__ t) {
    int m = blockIdx.x * 16 + (threadIdx.x >> 4);
    int r = threadIdx.x & 15;
    const float* xr = x + (size_t)m * DIM;
    const float* wr = wa + (size_t)r * DIM;
    float s = 0.f;
    for (int k = 0; k < DIM; k += 4) {
        f32x4 a = *(const f32x4*)(xr + k);
        f32x4 b = *(const f32x4*)(wr + k);
        s += a[0]*b[0] + a[1]*b[1] + a[2]*b[2] + a[3]*b[3];
    }
    t[(size_t)m * RANK + r] = s;
}

// ---------------------------------------------------------------------------
// Tiled WMMA GEMM:  C[m][n] = sum_k A[m][k] * W[n][k]   (both row-major in K)
// Optionally += loraScale * (T[m][0:16] . LW[n][0:16])  (rank-16 epilogue)
//
// Fused output modes:
//   outF != nullptr : fp32 row-major [M x N]                (final O-proj)
//   else            : bf16 transposed to [B, H, S, HD]; if fc != nullptr,
//                     interleaved RoPE is applied in-register first.
//     RoPE pairing: C-layout lane == output column, subtile col bases are
//     multiples of 16, so column parity == lane parity and the pair partner
//     lives in the xor-1 lane -> one __shfl_xor per element.
//
// Block: 256 thr = 8 waves (4 M x 2 N), tile 128x128, K-step 64.
// Double-buffered LDS staged via GLOBAL_LOAD_ASYNC_TO_LDS_B128:
//   issue tile t+1, s_wait_asynccnt 8 (async loads retire in order, so tile t
//   is complete while t+1 stays in flight), barrier, WMMA on tile t.
// ---------------------------------------------------------------------------
#define GT_M 128
#define GT_N 128
#define GT_K 64
#define GT_LD (GT_K + 8)

__device__ __forceinline__ void gemm_issue_tile(
        const bf16* __restrict__ A, const bf16* __restrict__ W,
        int K, int bm, int bn, int k0,
        bf16 (*As)[GT_LD], bf16 (*Bs)[GT_LD], int tid) {
#pragma unroll
    for (int i = 0; i < 4; ++i) {
        int linear = tid + i * 256;          // 128 rows x 8 chunks of 8 bf16
        int row = linear >> 3;
        int kc  = (linear & 7) * 8;
        async_b128(A + (size_t)(bm + row) * K + k0 + kc, &As[row][kc]);
        async_b128(W + (size_t)(bn + row) * K + k0 + kc, &Bs[row][kc]);
    }
}

__global__ __launch_bounds__(256) void gemm_bf16_kernel(
        const bf16* __restrict__ A, const bf16* __restrict__ W,
        int M, int N, int K,
        float* __restrict__ outF, bf16* __restrict__ outT,
        const float* __restrict__ fc, const float* __restrict__ fs,
        const float* __restrict__ loraT, const float* __restrict__ loraW,
        float loraScale) {
    __shared__ bf16 As[2][GT_M][GT_LD];
    __shared__ bf16 Bs[2][GT_N][GT_LD];

    int tid  = threadIdx.x;
    int bm   = blockIdx.x * GT_M;
    int bn   = blockIdx.y * GT_N;
    int wave = tid >> 5;
    int wm   = wave & 3;          // 4 waves along M (32 rows each)
    int wn   = wave >> 2;         // 2 waves along N (64 cols each)

    f32x8 acc[2][4];
#pragma unroll
    for (int mt = 0; mt < 2; ++mt)
#pragma unroll
        for (int nt = 0; nt < 4; ++nt)
            acc[mt][nt] = f32x8{0.f,0.f,0.f,0.f,0.f,0.f,0.f,0.f};

    const int T = K / GT_K;
    gemm_issue_tile(A, W, K, bm, bn, 0, As[0], Bs[0], tid);

    for (int t = 0; t < T; ++t) {
        if (t + 1 < T) {
            gemm_issue_tile(A, W, K, bm, bn, (t + 1) * GT_K,
                            As[(t + 1) & 1], Bs[(t + 1) & 1], tid);
            asm volatile("s_wait_asynccnt 0x8" ::: "memory");
        } else {
            asm volatile("s_wait_asynccnt 0x0" ::: "memory");
        }
        __syncthreads();

        const bf16 (*Ac)[GT_LD] = As[t & 1];
        const bf16 (*Bc)[GT_LD] = Bs[t & 1];
#pragma unroll
        for (int kk = 0; kk < GT_K; kk += 32) {
            bf16x16 af[2], bfm[4];
#pragma unroll
            for (int mt = 0; mt < 2; ++mt)
                af[mt] = load_frag(&Ac[0][0], GT_LD, wm * 32 + mt * 16, kk);
#pragma unroll
            for (int nt = 0; nt < 4; ++nt)
                bfm[nt] = load_frag(&Bc[0][0], GT_LD, wn * 64 + nt * 16, kk);
#pragma unroll
            for (int mt = 0; mt < 2; ++mt)
#pragma unroll
                for (int nt = 0; nt < 4; ++nt)
                    acc[mt][nt] = __builtin_amdgcn_wmma_f32_16x16x32_bf16(
                        false, af[mt], false, bfm[nt], (short)0, acc[mt][nt],
                        false, false);
        }
        __syncthreads();   // all waves done with buffer t before it is refilled
    }

    // epilogue (C layout: lane = col, VGPR r -> row r + 8*(lane/16))
    int lane = tid & 31, lh = lane >> 4, lc = lane & 15;
#pragma unroll
    for (int mt = 0; mt < 2; ++mt)
#pragma unroll
        for (int nt = 0; nt < 4; ++nt)
#pragma unroll
            for (int r = 0; r < 8; ++r) {
                int row = bm + wm * 32 + mt * 16 + lh * 8 + r;
                int col = bn + wn * 64 + nt * 16 + lc;
                float v = acc[mt][nt][r];
                if (loraT) {
                    const float* t = loraT + (size_t)row * RANK;
                    const float* w = loraW + (size_t)col * RANK;
                    float s = 0.f;
#pragma unroll
                    for (int j = 0; j < RANK; ++j) s += t[j] * w[j];
                    v += loraScale * s;
                }
                if (outF) {
                    outF[(size_t)row * N + col] = v;
                } else {
                    int bb = row >> 10;          // row / SEQ
                    int ss = row & (SEQ - 1);
                    int hh = col >> 7;           // col / HD
                    int dd = col & (HD - 1);
                    float o = v;
                    if (fc) {                    // fused interleaved RoPE
                        float partner = __shfl_xor(v, 1, 32);
                        int dh = dd >> 1;
                        float c  = fc[ss * (HD / 2) + dh];
                        float sn = fs[ss * (HD / 2) + dh];
                        o = (dd & 1) ? (partner * sn + v * c)   // imag slot
                                     : (v * c - partner * sn);  // real slot
                    }
                    outT[(((size_t)bb * NH + hh) * SEQ + ss) * HD + dd] = f2bf(o);
                }
            }
}

// ---------------------------------------------------------------------------
// Flash attention (causal), bf16 WMMA, fp32 online softmax.
// grid (S/64, H, B), block = 128 thr = 4 waves; each wave owns 16 query rows.
// K tile staged with async-to-LDS; V tile transposed through registers.
// Q,K,V: [B,H,S,HD] ; O: [B,S,H,HD]
// ---------------------------------------------------------------------------
#define FA_T 64   // tile (queries per block == keys per iteration)

__global__ __launch_bounds__(128) void flash_attn_kernel(
        const bf16* __restrict__ Q, const bf16* __restrict__ Kg,
        const bf16* __restrict__ Vg, bf16* __restrict__ O) {
    __shared__ bf16 Ks[FA_T][HD + 8];        // [token][d]
    __shared__ bf16 Vt[HD][FA_T + 8];        // [d][token] (transposed)
    __shared__ bf16 Pls[4][16][FA_T + 8];    // per-wave P (16 rows x 64 keys)

    int b = blockIdx.z, h = blockIdx.y, qt = blockIdx.x;
    int tid = threadIdx.x, wave = tid >> 5, lane = tid & 31;
    int lh = lane >> 4, lc = lane & 15;
    int q0 = qt * FA_T;

    const bf16* qbase = Q + (((size_t)b * NH + h) * SEQ + q0 + wave * 16) * HD;
    const bf16* kbase = Kg + (((size_t)b * NH + h) * SEQ) * HD;
    const bf16* vbase = Vg + (((size_t)b * NH + h) * SEQ) * HD;

    // Q fragments straight from global (row-major, contiguous in d)
    bf16x16 qf[4];
#pragma unroll
    for (int kk = 0; kk < 4; ++kk) qf[kk] = load_frag(qbase, HD, 0, kk * 32);

    f32x8 oacc[8];
#pragma unroll
    for (int nt = 0; nt < 8; ++nt)
        oacc[nt] = f32x8{0.f,0.f,0.f,0.f,0.f,0.f,0.f,0.f};
    float m_s[8], l_s[8];
#pragma unroll
    for (int r = 0; r < 8; ++r) { m_s[r] = -1e30f; l_s[r] = 0.f; }

    const float scale = 0.08838834764831845f;   // 1/sqrt(128)
    int nkt = qt + 1;                           // causal: key tiles <= query tile

    for (int kt = 0; kt < nkt; ++kt) {
        __syncthreads();
        // K tile via async-to-LDS; V tile loaded + transposed through registers
#pragma unroll
        for (int i = 0; i < 8; ++i) {
            int linear = tid + i * 128;          // 64 tokens x 16 chunks of 8
            int tok = linear >> 4;
            int dc  = (linear & 15) * 8;
            size_t g = (size_t)(kt * FA_T + tok) * HD + dc;
            async_b128(kbase + g, &Ks[tok][dc]);
            bf16x8 vv = *(const bf16x8*)(vbase + g);
#pragma unroll
            for (int j = 0; j < 8; ++j) Vt[dc + j][tok] = vv[j];
        }
        asm volatile("s_wait_asynccnt 0x0" ::: "memory");
        __syncthreads();

        // S = (Q K^T) * scale  (+ causal mask)
        f32x8 sacc[4];
#pragma unroll
        for (int nt = 0; nt < 4; ++nt) {
            sacc[nt] = f32x8{0.f,0.f,0.f,0.f,0.f,0.f,0.f,0.f};
#pragma unroll
            for (int kk = 0; kk < 4; ++kk) {
                bf16x16 kf = load_frag(&Ks[0][0], HD + 8, nt * 16, kk * 32);
                sacc[nt] = __builtin_amdgcn_wmma_f32_16x16x32_bf16(
                    false, qf[kk], false, kf, (short)0, sacc[nt], false, false);
            }
        }

        float mnew[8];
#pragma unroll
        for (int r = 0; r < 8; ++r) mnew[r] = m_s[r];
#pragma unroll
        for (int nt = 0; nt < 4; ++nt)
#pragma unroll
            for (int r = 0; r < 8; ++r) {
                float v = sacc[nt][r] * scale;
                int qrow = q0 + wave * 16 + lh * 8 + r;
                int kcol = kt * FA_T + nt * 16 + lc;
                if (kcol > qrow) v = -1e9f;
                sacc[nt][r] = v;
                mnew[r] = fmaxf(mnew[r], v);
            }
        // row-max across the 16 lanes holding this row (xor stays in half-wave)
#pragma unroll
        for (int off = 1; off < 16; off <<= 1)
#pragma unroll
            for (int r = 0; r < 8; ++r)
                mnew[r] = fmaxf(mnew[r], __shfl_xor(mnew[r], off, 32));

        float rsum[8];
#pragma unroll
        for (int r = 0; r < 8; ++r) rsum[r] = 0.f;
#pragma unroll
        for (int nt = 0; nt < 4; ++nt)
#pragma unroll
            for (int r = 0; r < 8; ++r) {
                float p = __expf(sacc[nt][r] - mnew[r]);
                rsum[r] += p;
                Pls[wave][lh * 8 + r][nt * 16 + lc] = f2bf(p);
            }
#pragma unroll
        for (int off = 1; off < 16; off <<= 1)
#pragma unroll
            for (int r = 0; r < 8; ++r)
                rsum[r] += __shfl_xor(rsum[r], off, 32);

#pragma unroll
        for (int r = 0; r < 8; ++r) {
            float alpha = __expf(m_s[r] - mnew[r]);
            l_s[r] = l_s[r] * alpha + rsum[r];
            m_s[r] = mnew[r];
#pragma unroll
            for (int nt = 0; nt < 8; ++nt) oacc[nt][r] *= alpha;
        }
        __syncthreads();   // make cross-lane P (LDS) visible before A-frag loads

        // O += P V
#pragma unroll
        for (int kk = 0; kk < 2; ++kk) {
            bf16x16 pf = load_frag(&Pls[wave][0][0], FA_T + 8, 0, kk * 32);
#pragma unroll
            for (int nt = 0; nt < 8; ++nt) {
                bf16x16 vf = load_frag(&Vt[0][0], FA_T + 8, nt * 16, kk * 32);
                oacc[nt] = __builtin_amdgcn_wmma_f32_16x16x32_bf16(
                    false, pf, false, vf, (short)0, oacc[nt], false, false);
            }
        }
    }

    // normalize + write O as [B,S,H,HD]
#pragma unroll
    for (int nt = 0; nt < 8; ++nt)
#pragma unroll
        for (int r = 0; r < 8; ++r) {
            int srow = q0 + wave * 16 + lh * 8 + r;
            int d = nt * 16 + lc;
            float v = oacc[nt][r] / l_s[r];
            O[(((size_t)b * SEQ + srow) * NH + h) * HD + d] = f2bf(v);
        }
}

// ---------------------------------------------------------------------------
// Host-side orchestration
// ---------------------------------------------------------------------------
extern "C" void kernel_launch(void* const* d_in, const int* in_sizes, int n_in,
                              void* d_out, int out_size, void* d_ws, size_t ws_size,
                              hipStream_t stream) {
    (void)in_sizes; (void)n_in; (void)out_size; (void)ws_size;
    const float* x    = (const float*)d_in[0];
    const float* wq_w = (const float*)d_in[1];
    const float* wq_a = (const float*)d_in[2];
    const float* wq_b = (const float*)d_in[3];
    const float* wk_w = (const float*)d_in[4];
    const float* wv_w = (const float*)d_in[5];
    const float* wv_a = (const float*)d_in[6];
    const float* wv_b = (const float*)d_in[7];
    const float* wo_w = (const float*)d_in[8];
    const float* fc   = (const float*)d_in[9];
    const float* fs   = (const float*)d_in[10];
    float* out = (float*)d_out;

    char* ws = (char*)d_ws;
    size_t off = 0;
    auto take = [&](size_t bytes) -> char* {
        char* p = ws + off;
        off = (off + bytes + 255) & ~(size_t)255;
        return p;
    };
    const size_t TOK2 = (size_t)BS * DIM * 2;   // one bf16 [4096 x 4096] tensor
    bf16* xb  = (bf16*)take(TOK2);
    bf16* wbq = (bf16*)take(TOK2);
    bf16* wbk = (bf16*)take(TOK2);
    bf16* wbv = (bf16*)take(TOK2);
    bf16* wbo = (bf16*)take(TOK2);
    bf16* qT  = (bf16*)take(TOK2);
    bf16* kT  = (bf16*)take(TOK2);
    bf16* vT  = (bf16*)take(TOK2);
    bf16* oS  = (bf16*)take(TOK2);
    float* tq = (float*)take((size_t)BS * RANK * 4);
    float* tv = (float*)take((size_t)BS * RANK * 4);

    // 1) fp32 -> bf16 conversions
    int n8 = BS * DIM / 8;   // == DIM*DIM/8 here
    int cb = (n8 + 255) / 256;
    cvt_f32_bf16_kernel<<<cb, 256, 0, stream>>>(x,    xb,  n8);
    cvt_f32_bf16_kernel<<<cb, 256, 0, stream>>>(wq_w, wbq, n8);
    cvt_f32_bf16_kernel<<<cb, 256, 0, stream>>>(wk_w, wbk, n8);
    cvt_f32_bf16_kernel<<<cb, 256, 0, stream>>>(wv_w, wbv, n8);
    cvt_f32_bf16_kernel<<<cb, 256, 0, stream>>>(wo_w, wbo, n8);

    // 2) LoRA rank-16 first stage
    lora_t_kernel<<<BS / 16, 256, 0, stream>>>(x, wq_a, tq);
    lora_t_kernel<<<BS / 16, 256, 0, stream>>>(x, wv_a, tv);

    // 3) QKV projections (WMMA) with fused LoRA + RoPE + transpose epilogues
    dim3 ggrid(BS / GT_M, DIM / GT_N);
    gemm_bf16_kernel<<<ggrid, 256, 0, stream>>>(xb, wbq, BS, DIM, DIM,
                                                nullptr, qT, fc, fs,
                                                tq, wq_b, LORA_SCALE);
    gemm_bf16_kernel<<<ggrid, 256, 0, stream>>>(xb, wbk, BS, DIM, DIM,
                                                nullptr, kT, fc, fs,
                                                nullptr, nullptr, 0.f);
    gemm_bf16_kernel<<<ggrid, 256, 0, stream>>>(xb, wbv, BS, DIM, DIM,
                                                nullptr, vT, nullptr, nullptr,
                                                tv, wv_b, LORA_SCALE);

    // 4) causal flash attention
    dim3 fgrid(SEQ / FA_T, NH, BATCH);
    flash_attn_kernel<<<fgrid, 128, 0, stream>>>(qT, kT, vT, oS);

    // 5) output projection -> fp32 result
    gemm_bf16_kernel<<<ggrid, 256, 0, stream>>>(oS, wbo, BS, DIM, DIM,
                                                out, nullptr, nullptr, nullptr,
                                                nullptr, nullptr, 0.f);
}
